// Attention_53979148976596
// MI455X (gfx1250) — compile-verified, matrix operand order checked by reading, not run on hardware
//
#include <hip/hip_runtime.h>

typedef __attribute__((ext_vector_type(16))) _Float16 v16h;
typedef __attribute__((ext_vector_type(8)))  float    v8f;
typedef int v4i_gcc __attribute__((vector_size(16)));

#define DIM   768
#define NH    12
#define HD    64
#define NSEQ  1024
#define BATCH 8
#define MROWS (BATCH * NSEQ)   // 8192
#define NQKV  (3 * DIM)        // 2304

#if defined(__has_builtin)
#if __has_builtin(__builtin_amdgcn_global_load_async_to_lds_b128)
#define HAVE_ASYNC_LDS 1
#endif
#endif

#ifdef HAVE_ASYNC_LDS
typedef __attribute__((address_space(1))) v4i_gcc* g4ptr;
typedef __attribute__((address_space(3))) v4i_gcc* l4ptr;
__device__ inline void async_cp16(void* lds_dst, const void* gsrc) {
  __builtin_amdgcn_global_load_async_to_lds_b128((g4ptr)(gsrc), (l4ptr)(lds_dst),
                                                 0, 0);
}
__device__ inline void async_wait0() {
  asm volatile("s_wait_asynccnt 0x0" ::: "memory");
}
#endif

// ---------------------------------------------------------------------------
// Fragment loaders (A-pattern per ISA 7.12.2; B assumed lane=column symmetric)
// Fragment = two 16B groups at +0B and +32B from base -> exactly 2 b128 loads.
// ---------------------------------------------------------------------------
__device__ inline v16h ld_frag_lds(const _Float16* lds, int row, int stride_h,
                                   int kbase, int lane) {
  union { v16h h; uint4 q[2]; } f;
  const int base = row * stride_h + kbase + ((lane >= 16) ? 8 : 0);
  f.q[0] = *(const uint4*)(lds + base);
  f.q[1] = *(const uint4*)(lds + base + 16);
  return f.h;
}

__device__ inline v16h ld_frag_glb(const _Float16* g, size_t base_half,
                                   int kbase, int lane) {
  union { v16h h; uint4 q[2]; } f;
  const size_t base = base_half + kbase + ((lane >= 16) ? 8 : 0);
  f.q[0] = *(const uint4*)(g + base);
  f.q[1] = *(const uint4*)(g + base + 16);
  return f.h;
}

__device__ inline v8f wmma32(v16h a, v16h b, v8f c) {
  return __builtin_amdgcn_wmma_f32_16x16x32_f16(false, a, false, b,
                                                (short)0, c, false, false);
}

__device__ inline unsigned pkh(float x, float y) {
  union { _Float16 h[2]; unsigned u; } t;
  t.h[0] = (_Float16)x; t.h[1] = (_Float16)y;
  return t.u;
}

__device__ inline float rmax16(float v) {
#pragma unroll
  for (int m = 8; m; m >>= 1) v = fmaxf(v, __shfl_xor(v, m, 16));
  return v;
}
__device__ inline float rsum16(float v) {
#pragma unroll
  for (int m = 8; m; m >>= 1) v += __shfl_xor(v, m, 16);
  return v;
}

// ---------------------------------------------------------------------------
// Kernel 1: convert qkv_w and proj_w to f16 (row-major [O,K] kept)
// ---------------------------------------------------------------------------
__global__ __launch_bounds__(256) void convert_w(const float* __restrict__ qkvw,
                                                 const float* __restrict__ projw,
                                                 _Float16* __restrict__ wq,
                                                 _Float16* __restrict__ wp) {
  const int t = blockIdx.x * 256 + threadIdx.x;
  if (t < NQKV * DIM) wq[t] = (_Float16)qkvw[t];
  else                wp[t - NQKV * DIM] = (_Float16)projw[t - NQKV * DIM];
}

// ---------------------------------------------------------------------------
// Kernel 2: bias in WMMA D-fragment order.
// biasD[h][qt (64)][kt (64)][lane (32)][r (8)]  (f16)
//   maps (row = qt*16 + r + (lane>=16?8:0), col = kt*16 + (lane&15)).
// One thread builds one 8-value lane chunk (16B contiguous store).
// ---------------------------------------------------------------------------
__global__ __launch_bounds__(256)
void build_bias(const int* __restrict__ rel_index,
                const float* __restrict__ rel_table,
                _Float16* __restrict__ biasD) {
  const int t = blockIdx.x * 256 + threadIdx.x;  // < 12*64*64*32
  const int lane = t & 31;
  const int kt = (t >> 5) & 63;
  const int qt = (t >> 11) & 63;
  const int h = t >> 17;
  const int kj = kt * 16 + (lane & 15);
  const int qbase = qt * 16 + ((lane >= 16) ? 8 : 0);
  union { uint4 q; _Float16 hx[8]; } out;
#pragma unroll
  for (int r = 0; r < 8; ++r) {
    const int idx = rel_index[(qbase + r) * NSEQ + kj];
    out.hx[r] = (_Float16)rel_table[idx * NH + h];
  }
  *(uint4*)(biasD + (size_t)t * 8) = out.q;
}

// ---------------------------------------------------------------------------
// Generic f16 WMMA GEMM: C[M,N] = A_f32[M,768] * W_f16[N,768]^T
// 128x128 block, 256 threads (8 waves, 2x4), wave tile 64x32, K-step 32.
// ---------------------------------------------------------------------------
#define GSTR 48  // LDS row stride in halves: 96B, 16B-aligned rows

template <int EP>
__global__ __launch_bounds__(256)
void gemm_f16(const float* __restrict__ A, const _Float16* __restrict__ W, int N,
              float* __restrict__ outF, const float* __restrict__ bias,
              _Float16* __restrict__ qb, _Float16* __restrict__ kb,
              _Float16* __restrict__ vtb,
              const float* __restrict__ q_bias, const float* __restrict__ v_bias) {
  alignas(16) __shared__ _Float16 lA[128 * GSTR];
  alignas(16) __shared__ _Float16 lB[128 * GSTR];

  const int tid = threadIdx.x;
  const int lane = tid & 31, w = tid >> 5;
  const int wm = (w >> 2) * 64, wn = (w & 3) * 32;
  const int M0 = blockIdx.y * 128, N0 = blockIdx.x * 128;
  const int ldrow = tid >> 1, ldpart = (tid & 1) * 16;  // 16 elems per thread

  v8f c[4][2];
#pragma unroll
  for (int mg = 0; mg < 4; ++mg)
#pragma unroll
    for (int ng = 0; ng < 2; ++ng) c[mg][ng] = {};

  const float*    aSrc = A + (size_t)(M0 + ldrow) * DIM + ldpart;
  const _Float16* wSrc = W + (size_t)(N0 + ldrow) * DIM + ldpart;

#pragma unroll 1
  for (int k0 = 0; k0 < DIM; k0 += 32) {
    __syncthreads();
    if (k0 + 32 < DIM) {  // prefetch next K-step tiles
      __builtin_prefetch(aSrc + k0 + 32, 0, 1);
      __builtin_prefetch(wSrc + k0 + 32, 0, 1);
    }
    {  // A fp32 -> f16 into LDS (needs conversion: stays on VGPR path)
      const float4* ap = (const float4*)(aSrc + k0);
      float4 f0 = ap[0], f1 = ap[1], f2 = ap[2], f3 = ap[3];
      unsigned* dst = (unsigned*)(lA + ldrow * GSTR + ldpart);
      dst[0] = pkh(f0.x, f0.y); dst[1] = pkh(f0.z, f0.w);
      dst[2] = pkh(f1.x, f1.y); dst[3] = pkh(f1.z, f1.w);
      dst[4] = pkh(f2.x, f2.y); dst[5] = pkh(f2.z, f2.w);
      dst[6] = pkh(f3.x, f3.y); dst[7] = pkh(f3.z, f3.w);
    }
    {  // W f16 tile: pure copy -> async global->LDS when available
      _Float16* dst = lB + ldrow * GSTR + ldpart;
#ifdef HAVE_ASYNC_LDS
      async_cp16(dst,     wSrc + k0);
      async_cp16(dst + 8, wSrc + k0 + 8);
      async_wait0();
#else
      const uint4* wp = (const uint4*)(wSrc + k0);
      ((uint4*)dst)[0] = wp[0];
      ((uint4*)dst)[1] = wp[1];
#endif
    }
    __syncthreads();

    v16h a[4], b[2];
#pragma unroll
    for (int mg = 0; mg < 4; ++mg)
      a[mg] = ld_frag_lds(lA, wm + mg * 16 + (lane & 15), GSTR, 0, lane);
#pragma unroll
    for (int ng = 0; ng < 2; ++ng)
      b[ng] = ld_frag_lds(lB, wn + ng * 16 + (lane & 15), GSTR, 0, lane);
#pragma unroll
    for (int mg = 0; mg < 4; ++mg)
#pragma unroll
      for (int ng = 0; ng < 2; ++ng) c[mg][ng] = wmma32(a[mg], b[ng], c[mg][ng]);
  }

  const int hi8 = (lane >= 16) ? 8 : 0;
#pragma unroll
  for (int mg = 0; mg < 4; ++mg)
#pragma unroll
    for (int ng = 0; ng < 2; ++ng)
#pragma unroll
      for (int r = 0; r < 8; ++r) {
        const int row = M0 + wm + mg * 16 + r + hi8;
        const int col = N0 + wn + ng * 16 + (lane & 15);
        const float val = c[mg][ng][r];
        if (EP == 0) {
          outF[(size_t)row * N + col] = val + bias[col];
        } else {
          const int bI = row >> 10, n = row & 1023;
          if (col < DIM) {
            const int hh = col >> 6, d = col & 63;
            qb[((size_t)(bI * NH + hh) * NSEQ + n) * HD + d] =
                (_Float16)((val + q_bias[col]) * 0.125f);
          } else if (col < 2 * DIM) {
            const int oo = col - DIM, hh = oo >> 6, d = oo & 63;
            kb[((size_t)(bI * NH + hh) * NSEQ + n) * HD + d] = (_Float16)val;
          } else {
            const int oo = col - 2 * DIM, hh = oo >> 6, d = oo & 63;
            vtb[((size_t)(bI * NH + hh) * HD + d) * NSEQ + n] =
                (_Float16)(val + v_bias[oo]);
          }
        }
      }
}

// ---------------------------------------------------------------------------
// Kernel 4: flash attention. 1 WG per (b,h,128-q-rows). 8 waves x 16 q rows.
// ---------------------------------------------------------------------------
__global__ __launch_bounds__(256)
void attn_kernel(const _Float16* __restrict__ qb, const _Float16* __restrict__ kb,
                 const _Float16* __restrict__ vtb,
                 const _Float16* __restrict__ biasD, float* __restrict__ aout) {
  alignas(16) __shared__ _Float16 lK[128 * 72];       // K tile  [key][64+pad]
  alignas(16) __shared__ _Float16 lV[64 * 136];       // V^T tile [d][128+pad]
  alignas(16) __shared__ _Float16 lP[8 * 16 * 136];   // per-wave P [16][128+pad]

  const int tid = threadIdx.x, lane = tid & 31, w = tid >> 5;
  const int qt = blockIdx.x & 7;
  const int h = (blockIdx.x >> 3) % NH;
  const int b = blockIdx.x / (8 * NH);
  const int q0 = qt * 128;
  const int hi8 = (lane >= 16) ? 8 : 0;
  const size_t headoff = (size_t)(b * NH + h) * NSEQ * HD;  // elements

  // per-thread staging sources/destinations
  const int krow = tid >> 1, kpart = (tid & 1) * 32;
  const int vd = tid >> 2, vpart = (tid & 3) * 32;
  const _Float16* kSrc = kb + headoff + (size_t)krow * HD + kpart;
  const _Float16* vSrc = vtb + ((size_t)(b * NH + h) * HD + vd) * NSEQ + vpart;
  _Float16* kDst = lK + krow * 72 + kpart;
  _Float16* vDst = lV + vd * 136 + vpart;

  // bias base for this wave's 16-row q tile (D-fragment order)
  const int qrow_tile = (q0 >> 4) + w;  // global 16-row tile index, 0..63
  const _Float16* bBase =
      biasD + ((((size_t)h * 64 + qrow_tile) * 64) * 256) + lane * 8;

  // Q fragments in registers (q already pre-scaled by 1/8)
  v16h aq[2];
  {
    const size_t base = headoff + (size_t)(q0 + w * 16 + (lane & 15)) * HD;
    aq[0] = ld_frag_glb(qb, base, 0, lane);
    aq[1] = ld_frag_glb(qb, base, 32, lane);
  }

  float rmax[8], runsum[8];
  v8f o[4];
#pragma unroll
  for (int r = 0; r < 8; ++r) { rmax[r] = -1e30f; runsum[r] = 0.f; }
#pragma unroll
  for (int dg = 0; dg < 4; ++dg) o[dg] = {};

#pragma unroll 1
  for (int j = 0; j < 8; ++j) {
    __syncthreads();
    {  // stage K block [128 keys][64] and V^T block [64 d][128 keys]
      const _Float16* ks = kSrc + (size_t)j * 128 * HD;
      const _Float16* vs = vSrc + j * 128;
#ifdef HAVE_ASYNC_LDS
      async_cp16(kDst,      ks);
      async_cp16(kDst + 8,  ks + 8);
      async_cp16(kDst + 16, ks + 16);
      async_cp16(kDst + 24, ks + 24);
      async_cp16(vDst,      vs);
      async_cp16(vDst + 8,  vs + 8);
      async_cp16(vDst + 16, vs + 16);
      async_cp16(vDst + 24, vs + 24);
      async_wait0();
#else
      const uint4* ks4 = (const uint4*)ks;
      uint4* kd4 = (uint4*)kDst;
      kd4[0] = ks4[0]; kd4[1] = ks4[1]; kd4[2] = ks4[2]; kd4[3] = ks4[3];
      const uint4* vs4 = (const uint4*)vs;
      uint4* vd4 = (uint4*)vDst;
      vd4[0] = vs4[0]; vd4[1] = vs4[1]; vd4[2] = vs4[2]; vd4[3] = vs4[3];
#endif
      if (j < 7) {  // prefetch next block
        __builtin_prefetch(ks + 128 * HD, 0, 1);
        __builtin_prefetch(vs + 128, 0, 1);
      }
    }
    __syncthreads();

    // S = q @ k^T (+bias): 8 n-frags of 16 keys, 2 K-steps over hd=64
    v8f s[8];
#pragma unroll
    for (int f = 0; f < 8; ++f) {
      v16h bk0 = ld_frag_lds(lK, f * 16 + (lane & 15), 72, 0, lane);
      v16h bk1 = ld_frag_lds(lK, f * 16 + (lane & 15), 72, 32, lane);
      v8f acc = {};
      acc = wmma32(aq[0], bk0, acc);
      acc = wmma32(aq[1], bk1, acc);
      s[f] = acc;
    }
    // bias add: one b128 load per 16x16 tile (D-fragment order)
#pragma unroll
    for (int f = 0; f < 8; ++f) {
      union { uint4 q; _Float16 hx[8]; } bb;
      bb.q = *(const uint4*)(bBase + (size_t)(j * 8 + f) * 256);
#pragma unroll
      for (int r = 0; r < 8; ++r) s[f][r] += (float)bb.hx[r];
    }

    // online softmax update (rows live across 16-lane halves)
#pragma unroll
    for (int r = 0; r < 8; ++r) {
      float bm = s[0][r];
#pragma unroll
      for (int f = 1; f < 8; ++f) bm = fmaxf(bm, s[f][r]);
      bm = rmax16(bm);
      const float nm = fmaxf(rmax[r], bm);
      const float corr = __expf(rmax[r] - nm);
      rmax[r] = nm;
      float rs = 0.f;
#pragma unroll
      for (int f = 0; f < 8; ++f) {
        const float p = __expf(s[f][r] - nm);
        s[f][r] = p;
        rs += p;
      }
      rs = rsum16(rs);
      runsum[r] = runsum[r] * corr + rs;
#pragma unroll
      for (int dg = 0; dg < 4; ++dg) o[dg][r] *= corr;
    }

    // P (D-layout) -> LDS -> A-layout
#pragma unroll
    for (int f = 0; f < 8; ++f)
#pragma unroll
      for (int r = 0; r < 8; ++r)
        lP[w * 2176 + (r + hi8) * 136 + f * 16 + (lane & 15)] =
            (_Float16)s[f][r];
    __syncthreads();

    // O += P @ V : K=128 keys in 4 steps of 32, 4 d-frags of 16
#pragma unroll
    for (int kk = 0; kk < 4; ++kk) {
      v16h pa = ld_frag_lds(lP, w * 16 + (lane & 15), 136, kk * 32, lane);
#pragma unroll
      for (int dg = 0; dg < 4; ++dg) {
        v16h bv = ld_frag_lds(lV, dg * 16 + (lane & 15), 136, kk * 32, lane);
        o[dg] = wmma32(pa, bv, o[dg]);
      }
    }
  }

  // normalize + write f32 [8192][768] at col h*64+d
  float inv[8];
#pragma unroll
  for (int r = 0; r < 8; ++r) inv[r] = 1.0f / runsum[r];
#pragma unroll
  for (int dg = 0; dg < 4; ++dg)
#pragma unroll
    for (int r = 0; r < 8; ++r) {
      const int gr = b * NSEQ + q0 + w * 16 + r + hi8;
      const int gc = h * HD + dg * 16 + (lane & 15);
      aout[(size_t)gr * DIM + gc] = o[dg][r] * inv[r];
    }
}

// ---------------------------------------------------------------------------
extern "C" void kernel_launch(void* const* d_in, const int* in_sizes, int n_in,
                              void* d_out, int out_size, void* d_ws,
                              size_t ws_size, hipStream_t stream) {
  const float* x        = (const float*)d_in[0];
  const float* qkv_w    = (const float*)d_in[1];
  const float* q_bias   = (const float*)d_in[2];
  const float* v_bias   = (const float*)d_in[3];
  const float* proj_w   = (const float*)d_in[4];
  const float* proj_b   = (const float*)d_in[5];
  const float* rel_table = (const float*)d_in[6];
  const int*   rel_index = (const int*)d_in[7];
  float* out = (float*)d_out;

  char* ws = (char*)d_ws;
  const size_t SZ_WQKV = (size_t)NQKV * DIM * 2;
  const size_t SZ_WPROJ = (size_t)DIM * DIM * 2;
  const size_t SZ_HEAD = (size_t)BATCH * NH * NSEQ * HD * 2;
  const size_t SZ_BIAS = (size_t)NH * NSEQ * NSEQ * 2;

  _Float16* wqkv  = (_Float16*)ws;              ws += SZ_WQKV;
  _Float16* wproj = (_Float16*)ws;              ws += SZ_WPROJ;
  _Float16* qbuf  = (_Float16*)ws;              ws += SZ_HEAD;
  _Float16* kbuf  = (_Float16*)ws;              ws += SZ_HEAD;
  _Float16* vtbuf = (_Float16*)ws;              ws += SZ_HEAD;
  _Float16* biasD = (_Float16*)ws;              ws += SZ_BIAS;
  float* aout = (float*)ws;  /* 8192*768*4 */

  convert_w<<<(NQKV * DIM + DIM * DIM) / 256, 256, 0, stream>>>(qkv_w, proj_w,
                                                                wqkv, wproj);
  build_bias<<<(NH * 64 * 64 * 32) / 256, 256, 0, stream>>>(rel_index,
                                                            rel_table, biasD);
  gemm_f16<1><<<dim3(NQKV / 128, MROWS / 128), 256, 0, stream>>>(
      x, wqkv, NQKV, nullptr, nullptr, qbuf, kbuf, vtbuf, q_bias, v_bias);
  attn_kernel<<<BATCH * NH * 8, 256, 0, stream>>>(qbuf, kbuf, vtbuf, biasD,
                                                  aout);
  gemm_f16<0><<<dim3(DIM / 128, MROWS / 128), 256, 0, stream>>>(
      aout, wproj, DIM, out, proj_b, nullptr, nullptr, nullptr, nullptr,
      nullptr);
}
